// CrossAttention_61229053772467
// MI455X (gfx1250) — compile-verified, matrix operand order checked by reading, not run on hardware
//
#include <hip/hip_runtime.h>
#include <hip/hip_bf16.h>
#include <stdint.h>

// ---------------------------------------------------------------------------
// CrossAttention for MI455X (gfx1250, wave32, WMMA, async-to-LDS).
// Pipeline: convert x/wq/wo/k/v -> f16 | qproj GEMM (WMMA, async dbl-buf) |
// flash attention (WMMA qk^T + online softmax + WMMA pv, async dbl-buf) |
// oproj GEMM (WMMA, async dbl-buf).
// ---------------------------------------------------------------------------

typedef __attribute__((ext_vector_type(16))) _Float16 v16h;
typedef __attribute__((ext_vector_type(8)))  float    v8f;

union Frag16 { v16h v; _Float16 h[16]; uint32_t u[8]; };

__device__ __forceinline__ v8f wmma_f16(v16h a, v16h b, v8f c) {
  return __builtin_amdgcn_wmma_f32_16x16x32_f16(false, a, false, b, (short)0, c,
                                                false, false);
}

// K index held by VGPR-pair j (element e=0) for a 16-bit A/B fragment,
// per CDNA5 ISA 7.12.2 (always even -> u32 pair loads are 4B aligned).
__device__ __forceinline__ int frag_k(int j, int half) {
  return ((j & 3) * 2) + half * 8 + ((j >> 2) * 16);
}

// 16B global->LDS copy: async (ASYNCcnt-tracked, no VGPR round trip) when the
// gfx1250 builtin is available, else synchronous fallback.
// Builtin signature (from probe diagnostic): param0 = int __vector(4) ptr in
// global address space; param1 = LDS side; then imm offset, imm cpol.
__device__ __forceinline__ void async_copy_b128(const void* g, void* l) {
#if __has_builtin(__builtin_amdgcn_global_load_async_to_lds_b128)
  typedef int v4i_gcc __attribute__((vector_size(16)));
  __builtin_amdgcn_global_load_async_to_lds_b128(
      (__attribute__((address_space(1))) v4i_gcc*)(void*)g,
      (__attribute__((address_space(3))) v4i_gcc*)l, 0, 0);
#else
  *(uint4*)l = *(const uint4*)g;
#endif
}

__device__ __forceinline__ void wait_async_all() {
  asm volatile("s_wait_asynccnt 0" ::: "memory");
}

constexpr int BB    = 8;
constexpr int T     = 1024;
constexpr int S     = 1500;
constexpr int S_PAD = 1504;   // multiple of 32
constexpr int D     = 1024;
constexpr int H     = 16;
constexpr int DH    = 64;
constexpr float SCALE = 0.125f; // 64^-0.5

// ---------------------------------------------------------------------------
// Plain f32 -> f16 convert (n multiple of 8).
// ---------------------------------------------------------------------------
__global__ void convert_f32_f16(const float* __restrict__ src,
                                _Float16* __restrict__ dst, size_t n) {
  size_t e0 = ((size_t)blockIdx.x * blockDim.x + threadIdx.x) * 8;
  if (e0 >= n) return;
  alignas(16) _Float16 o[8];
#pragma unroll
  for (int j = 0; j < 8; ++j) o[j] = (_Float16)src[e0 + j];
  *(uint4*)(dst + e0) = *(const uint4*)o;
}

// ---------------------------------------------------------------------------
// f32 -> f16 convert for K and V with zero padding to S_PAD rows.
// ---------------------------------------------------------------------------
__global__ void convert_kv_kernel(const float* __restrict__ k,
                                  const float* __restrict__ v,
                                  _Float16* __restrict__ kh,
                                  _Float16* __restrict__ vh) {
  const size_t total = (size_t)BB * S_PAD * D;
  size_t e0 = ((size_t)blockIdx.x * blockDim.x + threadIdx.x) * 8;
  if (e0 >= total) return;
  size_t b   = e0 / ((size_t)S_PAD * D);
  size_t rem = e0 - b * (size_t)S_PAD * D;
  int s = (int)(rem / D);
  int d = (int)(rem % D);
  alignas(16) _Float16 ko[8];
  alignas(16) _Float16 vo[8];
  if (s < S) {
    const float* ks = k + ((size_t)b * S + s) * D + d;
    const float* vs = v + ((size_t)b * S + s) * D + d;
#pragma unroll
    for (int j = 0; j < 8; ++j) { ko[j] = (_Float16)ks[j]; vo[j] = (_Float16)vs[j]; }
  } else {
#pragma unroll
    for (int j = 0; j < 8; ++j) { ko[j] = (_Float16)0.f; vo[j] = (_Float16)0.f; }
  }
  *(uint4*)(kh + e0) = *(const uint4*)ko;
  *(uint4*)(vh + e0) = *(const uint4*)vo;
}

// ---------------------------------------------------------------------------
// C[M,N] = (A[M,K] @ W[N,K]^T + bias[N]) * alpha   (torch Linear semantics)
// A and W are f16. Block = 256 threads (8 waves), tile 128(M) x 64(N),
// K-step 32, async double-buffered LDS staging. Each wave: one 16-row strip,
// 4 WMMA C tiles (16x64).
// ---------------------------------------------------------------------------
template <typename OT>
__global__ void gemm_bias_wmma(const _Float16* __restrict__ A,
                               const _Float16* __restrict__ W,
                               const float* __restrict__ bias,
                               OT* __restrict__ Cout,
                               int M, int N, int K, float alpha) {
  __shared__ _Float16 Alds[2][128][40];   // 80B rows (16B aligned)
  __shared__ _Float16 Wlds[2][64][40];

  const int tid  = threadIdx.x;
  const int wave = tid >> 5;
  const int lane = tid & 31;
  const int half = lane >> 4;
  const int idx  = lane & 15;
  const int m0   = blockIdx.x * 128;
  const int n0   = blockIdx.y * 64;

  // staging coordinates
  const int arow  = tid >> 1;          // 0..127
  const int acoff = (tid & 1) * 16;    // 16 halves = 2 x b128
  const int wrow  = tid >> 2;          // 0..63
  const int wcoff = (tid & 3) * 8;     // 8 halves = 1 x b128

  auto issue_tiles = [&](int buf, int k0) {
    const _Float16* asrc = A + (size_t)(m0 + arow) * K + k0 + acoff;
    async_copy_b128(asrc,     &Alds[buf][arow][acoff]);
    async_copy_b128(asrc + 8, &Alds[buf][arow][acoff + 8]);
    const _Float16* wsrc = W + (size_t)(n0 + wrow) * K + k0 + wcoff;
    async_copy_b128(wsrc, &Wlds[buf][wrow][wcoff]);
  };

  v8f acc[4] = {};

  const int nsteps = K / 32;
  issue_tiles(0, 0);
  for (int ks = 0; ks < nsteps; ++ks) {
    const int buf = ks & 1;
    wait_async_all();
    __syncthreads();
    if (ks + 1 < nsteps) issue_tiles(buf ^ 1, (ks + 1) * 32);

    Frag16 a;
#pragma unroll
    for (int j = 0; j < 8; ++j)
      a.u[j] = *(const uint32_t*)&Alds[buf][wave * 16 + idx][frag_k(j, half)];

    Frag16 b0, b1, b2, b3;
#pragma unroll
    for (int j = 0; j < 8; ++j) {
      const int kk = frag_k(j, half);
      b0.u[j] = *(const uint32_t*)&Wlds[buf][ 0 + idx][kk];
      b1.u[j] = *(const uint32_t*)&Wlds[buf][16 + idx][kk];
      b2.u[j] = *(const uint32_t*)&Wlds[buf][32 + idx][kk];
      b3.u[j] = *(const uint32_t*)&Wlds[buf][48 + idx][kk];
    }
    acc[0] = wmma_f16(a.v, b0.v, acc[0]);
    acc[1] = wmma_f16(a.v, b1.v, acc[1]);
    acc[2] = wmma_f16(a.v, b2.v, acc[2]);
    acc[3] = wmma_f16(a.v, b3.v, acc[3]);
  }

#pragma unroll
  for (int dt = 0; dt < 4; ++dt) {
    const int n  = n0 + dt * 16 + idx;
    const float bv = bias[n];
#pragma unroll
    for (int r = 0; r < 8; ++r) {
      const int m = m0 + wave * 16 + half * 8 + r;
      float val = (acc[dt][r] + bv) * alpha;
      Cout[(size_t)m * N + n] = (OT)val;
    }
  }
}

// ---------------------------------------------------------------------------
// Flash attention core. Block = 8 waves; block handles (b, h, 128 T rows);
// each wave owns a 16-row q strip. S loop in chunks of 32 (one WMMA K step),
// K/V chunks async double-buffered into LDS.
// ---------------------------------------------------------------------------
__global__ void attn_flash_wmma(const _Float16* __restrict__ qh,
                                const _Float16* __restrict__ kh,
                                const _Float16* __restrict__ vh,
                                _Float16* __restrict__ oh) {
  __shared__ _Float16 Klds[2][32][72];     // [s_local][d], 144B rows
  __shared__ _Float16 Vlds[2][32][72];     // [s_local][d], row-major
  __shared__ _Float16 Plds[8][16][34];     // per-wave P transpose bounce

  const int tid  = threadIdx.x;
  const int wave = tid >> 5;
  const int lane = tid & 31;
  const int half = lane >> 4;
  const int idx  = lane & 15;
  const int b = blockIdx.z;
  const int h = blockIdx.y;
  const int t0 = blockIdx.x * 128 + wave * 16;

  // Q fragments (A layout): rows t0..t0+15, k over d (0..31) and (32..63)
  Frag16 qa0, qa1;
  {
    const _Float16* qrow = qh + ((size_t)(b * T + t0 + idx)) * D + h * DH;
#pragma unroll
    for (int j = 0; j < 8; ++j) {
      const int kk = frag_k(j, half);
      qa0.u[j] = *(const uint32_t*)(qrow + kk);
      qa1.u[j] = *(const uint32_t*)(qrow + 32 + kk);
    }
  }

  v8f acc[4] = {};
  float mrow[8], lrow[8];
#pragma unroll
  for (int r = 0; r < 8; ++r) { mrow[r] = -3.0e38f; lrow[r] = 0.f; }

  const _Float16* kbase = kh + (size_t)b * S_PAD * D + h * DH;
  const _Float16* vbase = vh + (size_t)b * S_PAD * D + h * DH;

  const int srow = tid >> 3;   // 0..31
  const int sseg = tid & 7;    // 8 halves = 1 x b128 each for K and V

  auto issue_chunk = [&](int buf, int c) {
    const int s0 = c * 32;
    const _Float16* kr = kbase + (size_t)(s0 + srow) * D + sseg * 8;
    const _Float16* vr = vbase + (size_t)(s0 + srow) * D + sseg * 8;
    async_copy_b128(kr, &Klds[buf][srow][sseg * 8]);
    async_copy_b128(vr, &Vlds[buf][srow][sseg * 8]);
  };

  const int NCH = S_PAD / 32;
  issue_chunk(0, 0);
  for (int c = 0; c < NCH; ++c) {
    const int buf = c & 1;
    const int s0 = c * 32;
    wait_async_all();
    __syncthreads();
    if (c + 1 < NCH) issue_chunk(buf ^ 1, c + 1);

    // scores: two 16x16 tiles (cols s0+idx, s0+16+idx), K = DH = 2 WMMA steps
    v8f sc0 = {}, sc1 = {};
    {
      Frag16 kb;
#pragma unroll
      for (int j = 0; j < 8; ++j)
        kb.u[j] = *(const uint32_t*)&Klds[buf][idx][frag_k(j, half)];
      sc0 = wmma_f16(qa0.v, kb.v, sc0);
#pragma unroll
      for (int j = 0; j < 8; ++j)
        kb.u[j] = *(const uint32_t*)&Klds[buf][idx][32 + frag_k(j, half)];
      sc0 = wmma_f16(qa1.v, kb.v, sc0);
#pragma unroll
      for (int j = 0; j < 8; ++j)
        kb.u[j] = *(const uint32_t*)&Klds[buf][16 + idx][frag_k(j, half)];
      sc1 = wmma_f16(qa0.v, kb.v, sc1);
#pragma unroll
      for (int j = 0; j < 8; ++j)
        kb.u[j] = *(const uint32_t*)&Klds[buf][16 + idx][32 + frag_k(j, half)];
      sc1 = wmma_f16(qa1.v, kb.v, sc1);
    }

    // mask padded columns (only last chunk)
    if (s0 + idx >= S) {
#pragma unroll
      for (int r = 0; r < 8; ++r) sc0[r] = -1.0e30f;
    }
    if (s0 + 16 + idx >= S) {
#pragma unroll
      for (int r = 0; r < 8; ++r) sc1[r] = -1.0e30f;
    }

    // online softmax; rows live across 16-lane groups (C layout)
#pragma unroll
    for (int r = 0; r < 8; ++r) {
      float v0 = sc0[r], v1 = sc1[r];
      float mx = fmaxf(v0, v1);
      mx = fmaxf(mx, __shfl_xor(mx, 1, 16));
      mx = fmaxf(mx, __shfl_xor(mx, 2, 16));
      mx = fmaxf(mx, __shfl_xor(mx, 4, 16));
      mx = fmaxf(mx, __shfl_xor(mx, 8, 16));
      const float mnew = fmaxf(mrow[r], mx);
      const float corr = __expf(mrow[r] - mnew);
      const float p0 = __expf(v0 - mnew);
      const float p1 = __expf(v1 - mnew);
      float sm = p0 + p1;
      sm += __shfl_xor(sm, 1, 16);
      sm += __shfl_xor(sm, 2, 16);
      sm += __shfl_xor(sm, 4, 16);
      sm += __shfl_xor(sm, 8, 16);
      lrow[r] = lrow[r] * corr + sm;
      mrow[r] = mnew;
      acc[0][r] *= corr; acc[1][r] *= corr; acc[2][r] *= corr; acc[3][r] *= corr;
      // C layout -> LDS (row m = half*8 + r, cols idx / 16+idx)
      Plds[wave][half * 8 + r][idx]      = (_Float16)p0;
      Plds[wave][half * 8 + r][16 + idx] = (_Float16)p1;
    }
    asm volatile("s_wait_dscnt 0" ::: "memory");  // in-wave LDS RAW for P bounce

    // P as A fragment (16x32)
    Frag16 pa;
#pragma unroll
    for (int j = 0; j < 8; ++j)
      pa.u[j] = *(const uint32_t*)&Plds[wave][idx][frag_k(j, half)];

    // acc += P @ V_chunk  (4 d-tiles, K = 32); V read row-major [k][n]
#pragma unroll
    for (int dt = 0; dt < 4; ++dt) {
      Frag16 vb;
#pragma unroll
      for (int j = 0; j < 8; ++j) {
        const int kk = frag_k(j, half);
        vb.h[2 * j]     = Vlds[buf][kk][dt * 16 + idx];
        vb.h[2 * j + 1] = Vlds[buf][kk + 1][dt * 16 + idx];
      }
      acc[dt] = wmma_f16(pa.v, vb.v, acc[dt]);
    }
  }

  // epilogue: normalize and store f16 attn output in [B,T,D] layout
#pragma unroll
  for (int dt = 0; dt < 4; ++dt) {
#pragma unroll
    for (int r = 0; r < 8; ++r) {
      const float o = acc[dt][r] / lrow[r];
      const int t = t0 + half * 8 + r;
      oh[((size_t)(b * T + t)) * D + h * DH + dt * 16 + idx] = (_Float16)o;
    }
  }
}

// ---------------------------------------------------------------------------
extern "C" void kernel_launch(void* const* d_in, const int* in_sizes, int n_in,
                              void* d_out, int out_size, void* d_ws, size_t ws_size,
                              hipStream_t stream) {
  const float* x  = (const float*)d_in[0];
  const float* k  = (const float*)d_in[1];
  const float* v  = (const float*)d_in[2];
  const float* wq = (const float*)d_in[3];
  const float* bq = (const float*)d_in[4];
  const float* wo = (const float*)d_in[5];
  const float* bo = (const float*)d_in[6];
  float* out = (float*)d_out;

  char* ws = (char*)d_ws;
  const size_t qh_sz = (size_t)BB * T * D * sizeof(_Float16);        // 16 MiB
  const size_t kv_sz = (size_t)BB * S_PAD * D * sizeof(_Float16);    // ~23.5 MiB
  const size_t w_sz  = (size_t)D * D * sizeof(_Float16);             // 2 MiB
  size_t off = 0;
  _Float16* qh  = (_Float16*)(ws + off); off += qh_sz;
  _Float16* kh  = (_Float16*)(ws + off); off += kv_sz;
  _Float16* vh  = (_Float16*)(ws + off); off += kv_sz;
  _Float16* oh  = (_Float16*)(ws + off); off += qh_sz;
  _Float16* xh  = (_Float16*)(ws + off); off += qh_sz;
  _Float16* wqh = (_Float16*)(ws + off); off += w_sz;
  _Float16* woh = (_Float16*)(ws + off); off += w_sz;

  // 1) one-time f16 conversions
  {
    const size_t nx = (size_t)BB * T * D;
    convert_f32_f16<<<(int)((nx / 8 + 255) / 256), 256, 0, stream>>>(x, xh, nx);
    const size_t nw = (size_t)D * D;
    convert_f32_f16<<<(int)((nw / 8 + 255) / 256), 256, 0, stream>>>(wq, wqh, nw);
    convert_f32_f16<<<(int)((nw / 8 + 255) / 256), 256, 0, stream>>>(wo, woh, nw);
    const size_t nkv = (size_t)BB * S_PAD * D;
    convert_kv_kernel<<<(int)((nkv / 8 + 255) / 256), 256, 0, stream>>>(k, v, kh, vh);
  }
  // 2) q = (x @ wq^T + bq) * SCALE  -> f16
  gemm_bias_wmma<_Float16><<<dim3((BB * T) / 128, D / 64), 256, 0, stream>>>(
      xh, wqh, bq, qh, BB * T, D, D, SCALE);
  // 3) flash attention -> f16 [B,T,D]
  attn_flash_wmma<<<dim3(T / 128, H, BB), 256, 0, stream>>>(qh, kh, vh, oh);
  // 4) out = oh @ wo^T + bo -> f32
  gemm_bias_wmma<float><<<dim3((BB * T) / 128, D / 64), 256, 0, stream>>>(
      oh, woh, bo, out, BB * T, D, D, 1.0f);
}
